// LSTM_67731634258775
// MI455X (gfx1250) — matrix operand order repaired and checked by reading, not compiled, not run on hardware
//
#include <hip/hip_runtime.h>
#include <stdint.h>

#define B_    8192
#define D_    8000
#define H_    100
#define G4H_  400
#define V_    8000
#define KPAD_ 128
#define VPAD_ 8064   // 63 * 128

typedef unsigned int       u32;
typedef unsigned long long u64;
typedef unsigned short     u16;

typedef __attribute__((ext_vector_type(16))) __bf16 v16bf;
typedef __attribute__((ext_vector_type(8)))  float  v8f;
typedef __attribute__((ext_vector_type(4)))  float  f32x4;
typedef __attribute__((ext_vector_type(4)))  u32    u32x4;
typedef __attribute__((ext_vector_type(8)))  int    i32x8;
typedef __attribute__((ext_vector_type(4)))  int    i32x4;

union ABf { v16bf v; u32x4 q[2]; };

// ---------------- workspace layout (bytes, each region 256B aligned) ----------------
static constexpr size_t OFF_UHI   = 0;
static constexpr size_t OFF_ULO   = OFF_UHI   + (size_t)G4H_  * D_    * 2;   // 6.4 MB each
static constexpr size_t OFF_WHI   = OFF_ULO   + (size_t)G4H_  * D_    * 2;
static constexpr size_t OFF_WLO   = OFF_WHI   + (size_t)G4H_  * KPAD_ * 2;
static constexpr size_t OFF_VHI   = OFF_WLO   + (size_t)G4H_  * KPAD_ * 2;
static constexpr size_t OFF_VLO   = OFF_VHI   + (size_t)VPAD_ * KPAD_ * 2;
static constexpr size_t OFF_HHI   = OFF_VLO   + (size_t)VPAD_ * KPAD_ * 2;
static constexpr size_t OFF_HLO   = OFF_HHI   + (size_t)B_    * KPAD_ * 2;
static constexpr size_t OFF_GATES = OFF_HLO   + (size_t)B_    * KPAD_ * 2;
static constexpr size_t OFF_LSE   = OFF_GATES + (size_t)B_    * G4H_  * 4;
// total ~33 MB

// ---------------- helpers ----------------
// native fp32 -> bf16 (RNE) split; compiler lowers to packed v_cvt on gfx1250
__device__ inline void cvt16(const float* f, ABf& hi, ABf& lo) {
#pragma unroll
    for (int e = 0; e < 16; ++e) {
        __bf16 h = (__bf16)f[e];
        hi.v[e] = h;
        lo.v[e] = (__bf16)(f[e] - (float)h);
    }
}

__device__ inline v8f wmma_bf16(v16bf a, v16bf b, v8f c) {
#if defined(__HIP_DEVICE_COMPILE__)
    return __builtin_amdgcn_wmma_f32_16x16x32_bf16(false, a, false, b, (short)0, c, false, false);
#else
    return c;
#endif
}
// split-bf16 emulated fp32: hi*hi + lo*hi + hi*lo
__device__ inline v8f wmma3(const ABf& ah, const ABf& al, const ABf& bh, const ABf& bl, v8f c) {
    c = wmma_bf16(ah.v, bh.v, c);
    c = wmma_bf16(al.v, bh.v, c);
    c = wmma_bf16(ah.v, bl.v, c);
    return c;
}

#if defined(__HIP_DEVICE_COMPILE__) && __has_builtin(__builtin_amdgcn_tensor_load_to_lds)
#define HAS_TDM 1
#else
#define HAS_TDM 0
#endif

#if HAS_TDM
// 2D TDM load: tile 32 cols (fp32) x 128 rows into LDS, LDS rows padded to 36 dwords
// (pad_interval code 4 = every 32 dwords, pad_amount code 3 = 4 dwords).
__device__ inline void tdm_load_x(u32 lds_off, u64 ga, u32 rows_left, u32 cols_left) {
    u32x4 g0;
    g0[0] = 1u;                                        // count=1, user descriptor
    g0[1] = lds_off;                                   // lds_addr
    g0[2] = (u32)ga;                                   // global_addr[31:0]
    g0[3] = (u32)((ga >> 32) & 0x1FFFFFFu) | (2u << 30); // addr[56:32] | type=2
    i32x8 g1;
    g1[0] = (int)((2u << 16) | (1u << 20) | (4u << 22) | (3u << 25)); // data_size=4B, pad_en, int=32dw, amt=4dw
    g1[1] = (int)((cols_left & 0xFFFFu) << 16);                        // tensor_dim0[15:0]
    g1[2] = (int)(((cols_left >> 16) & 0xFFFFu) | ((rows_left & 0xFFFFu) << 16)); // dim0 hi | dim1 lo
    g1[3] = (int)(((rows_left >> 16) & 0xFFFFu) | (32u << 16));        // dim1 hi | tile_dim0=32
    g1[4] = (int)(128u);                                               // tile_dim1=128, tile_dim2=0
    g1[5] = (int)((u32)D_);                                            // tensor_dim0_stride lo
    g1[6] = 0;                                                         // stride hi | dim1_stride lo
    g1[7] = 0;
    i32x4 gz4 = {0, 0, 0, 0};
    i32x8 gz8 = {0, 0, 0, 0, 0, 0, 0, 0};
    // clang-23 / therock-10.0 arity: (g0, g1, g2, g3, extra, cpol)
    __builtin_amdgcn_tensor_load_to_lds(g0, g1, gz4, gz4, gz8, 0);
}
#endif

// ---------------- pack kernels (fp32 -> bf16 hi/lo, fused/padded layouts) ----------------
__global__ __launch_bounds__(256) void pack_U_kernel(
    const float* __restrict__ Ui, const float* __restrict__ Uf,
    const float* __restrict__ Uo, const float* __restrict__ Ug,
    __bf16* __restrict__ Uhi, __bf16* __restrict__ Ulo)
{
    size_t idx = (size_t)blockIdx.x * 256 + threadIdx.x;
    if (idx >= (size_t)G4H_ * D_) return;
    int n = (int)(idx / D_), k = (int)(idx % D_);
    int g = n / H_, j = n % H_;
    const float* src = (g == 0) ? Ui : (g == 1) ? Uf : (g == 2) ? Uo : Ug;
    float v = src[(size_t)j * D_ + k];
    __bf16 h = (__bf16)v;
    Uhi[idx] = h;
    Ulo[idx] = (__bf16)(v - (float)h);
}

__global__ __launch_bounds__(256) void pack_W_kernel(
    const float* __restrict__ Wi, const float* __restrict__ Wf,
    const float* __restrict__ Wo, const float* __restrict__ Wg,
    __bf16* __restrict__ Whi, __bf16* __restrict__ Wlo)
{
    int idx = blockIdx.x * 256 + threadIdx.x;
    if (idx >= G4H_ * KPAD_) return;
    int n = idx / KPAD_, k = idx % KPAD_;
    int g = n / H_, j = n % H_;
    float v = 0.f;
    if (k < H_) {
        const float* src = (g == 0) ? Wi : (g == 1) ? Wf : (g == 2) ? Wo : Wg;
        v = src[(size_t)j * H_ + k];
    }
    __bf16 h = (__bf16)v;
    Whi[idx] = h;
    Wlo[idx] = (__bf16)(v - (float)h);
}

__global__ __launch_bounds__(256) void pack_V_kernel(
    const float* __restrict__ Vw, __bf16* __restrict__ Vhi, __bf16* __restrict__ Vlo)
{
    size_t idx = (size_t)blockIdx.x * 256 + threadIdx.x;
    if (idx >= (size_t)VPAD_ * KPAD_) return;
    int n = (int)(idx / KPAD_), k = (int)(idx % KPAD_);
    float v = (n < V_ && k < H_) ? Vw[(size_t)n * H_ + k] : 0.f;
    __bf16 h = (__bf16)v;
    Vhi[idx] = h;
    Vlo[idx] = (__bf16)(v - (float)h);
}

// ---------------- GEMM 1: gates = x @ U^T + hidden @ W^T ----------------
// grid (5, 64): blockIdx.x = 80-col N-slab, blockIdx.y = 128-row M-slab. 8 wave32 waves,
// wave w owns rows w*16..w*16+15 and 5 N-tiles (5 x v8f accumulators).
__global__ __launch_bounds__(256) void gates_gemm_kernel(
    const float* __restrict__ x, const float* __restrict__ hidden,
    const __bf16* __restrict__ Uhi, const __bf16* __restrict__ Ulo,
    const __bf16* __restrict__ Whi, const __bf16* __restrict__ Wlo,
    float* __restrict__ gates)
{
    __shared__ __align__(16) float xs[2][128][36];   // 36.9 KB, rows padded 32->36 dwords
    const int tid  = threadIdx.x;
    const int wave = tid >> 5, lane = tid & 31;
    const int kh   = lane >> 4, l16 = lane & 15;     // kh: K-half of this lane
    const int row_base = blockIdx.y * 128;
    const int ncb      = blockIdx.x * 80;
    const int mrow     = row_base + wave * 16 + l16;

    v8f acc[5] = {};

    const int KT = D_ / 32;   // 250
#if HAS_TDM
    const u64 xbase = (u64)(uintptr_t)(const void*)(x + (size_t)row_base * D_);
    int cur = 0;
    if (wave == 0)
        tdm_load_x((u32)(uintptr_t)&xs[0][0][0], xbase, (u32)(B_ - row_base), (u32)D_);
#endif

    for (int kt = 0; kt < KT; ++kt) {
        const int k0 = kt * 32;
#if HAS_TDM
        if (wave == 0) __builtin_amdgcn_s_wait_tensorcnt(0);
        __syncthreads();
        if (wave == 0 && kt + 1 < KT)
            tdm_load_x((u32)(uintptr_t)&xs[cur ^ 1][0][0], xbase + (u64)(k0 + 32) * 4,
                       (u32)(B_ - row_base), (u32)(D_ - (k0 + 32)));
        const int rb = cur;
#else
        __syncthreads();
        for (int i = tid; i < 1024; i += 256) {      // 128 rows x 8 f32x4
            int r = i >> 3, c4 = (i & 7) << 2;
            *(f32x4*)&xs[0][r][c4] =
                *(const f32x4*)(x + (size_t)(row_base + r) * D_ + k0 + c4);
        }
        if (kt + 1 < KT)                             // pull next slab toward L2
            __builtin_prefetch(x + (size_t)(row_base + (tid >> 1)) * D_ + k0 + 32, 0, 1);
        __syncthreads();
        const int rb = 0;
#endif
        // A fragment: row (wave*16+l16), one contiguous 16-float K-chunk per lane, split to bf16 hi/lo
        const float* rp = &xs[rb][wave * 16 + l16][0];
        float fa[16];
        f32x4* fv = (f32x4*)fa;
        fv[0] = *(const f32x4*)(rp + kh * 16);
        fv[1] = *(const f32x4*)(rp + kh * 16 + 4);
        fv[2] = *(const f32x4*)(rp + kh * 16 + 8);
        fv[3] = *(const f32x4*)(rp + kh * 16 + 12);
        ABf ah, al;
        cvt16(fa, ah, al);

#pragma unroll
        for (int t = 0; t < 5; ++t) {
            const int n = ncb + t * 16 + l16;
            const size_t bo = (size_t)n * D_ + k0 + kh * 16;
            ABf bh, bl;
            bh.v = *(const v16bf*)(Uhi + bo);
            bl.v = *(const v16bf*)(Ulo + bo);
            acc[t] = wmma3(ah, al, bh, bl, acc[t]);
        }
#if HAS_TDM
        cur ^= 1;
#endif
    }

    // hidden @ W^T, K = 100 zero-padded to 128
#pragma unroll
    for (int kt = 0; kt < 4; ++kt) {
        const int k0 = kt * 32;
        float fa[16];
#pragma unroll
        for (int e = 0; e < 8; ++e) {
            int ka = k0 + kh * 16 + e;
            int kb = k0 + kh * 16 + 8 + e;
            fa[e]     = (ka < H_) ? hidden[(size_t)mrow * H_ + ka] : 0.f;
            fa[8 + e] = (kb < H_) ? hidden[(size_t)mrow * H_ + kb] : 0.f;
        }
        ABf ah, al;
        cvt16(fa, ah, al);
#pragma unroll
        for (int t = 0; t < 5; ++t) {
            const int n = ncb + t * 16 + l16;
            const size_t bo = (size_t)n * KPAD_ + k0 + kh * 16;
            ABf bh, bl;
            bh.v = *(const v16bf*)(Whi + bo);
            bl.v = *(const v16bf*)(Wlo + bo);
            acc[t] = wmma3(ah, al, bh, bl, acc[t]);
        }
    }

    // store: C/D layout -> element (M = kh*8+r, N = l16)
#pragma unroll
    for (int t = 0; t < 5; ++t) {
        const int col = ncb + t * 16 + l16;
#pragma unroll
        for (int r = 0; r < 8; ++r) {
            const int row = row_base + wave * 16 + kh * 8 + r;
            gates[(size_t)row * G4H_ + col] = acc[t][r];
        }
    }
}

// ---------------- LSTM elementwise + bf16 hi/lo pack of h_t ----------------
__global__ __launch_bounds__(256) void lstm_kernel(
    const float* __restrict__ gates, const float* __restrict__ c_in,
    const float* __restrict__ Ui_b, const float* __restrict__ Uf_b,
    const float* __restrict__ Uo_b, const float* __restrict__ Ug_b,
    const float* __restrict__ Wi_b, const float* __restrict__ Wf_b,
    const float* __restrict__ Wo_b, const float* __restrict__ Wg_b,
    float* __restrict__ h_out, float* __restrict__ c_out,
    __bf16* __restrict__ Hhi, __bf16* __restrict__ Hlo)
{
    size_t idx = (size_t)blockIdx.x * 256 + threadIdx.x;   // over B * KPAD
    if (idx >= (size_t)B_ * KPAD_) return;
    int b = (int)(idx / KPAD_), j = (int)(idx % KPAD_);
    if (j < H_) {
        const size_t base = (size_t)b * G4H_;
        float gi = gates[base + j]            + Ui_b[j] + Wi_b[j];
        float gf = gates[base + H_ + j]       + Uf_b[j] + Wf_b[j];
        float go = gates[base + 2 * H_ + j]   + Uo_b[j] + Wo_b[j];
        float gg = gates[base + 3 * H_ + j]   + Ug_b[j] + Wg_b[j];
        float it = 1.f / (1.f + __expf(-gi));
        float ft = 1.f / (1.f + __expf(-gf));
        float ot = 1.f / (1.f + __expf(-go));
        float gt = tanhf(gg);
        float ct = c_in[(size_t)b * H_ + j] * ft + gt * it;
        float ht = tanhf(ct) * ot;
        h_out[(size_t)b * H_ + j] = ht;
        c_out[(size_t)b * H_ + j] = ct;
        __bf16 hh = (__bf16)ht;
        Hhi[idx] = hh;
        Hlo[idx] = (__bf16)(ht - (float)hh);
    } else {
        Hhi[idx] = (__bf16)0.f;
        Hlo[idx] = (__bf16)0.f;
    }
}

// ---------------- GEMM 3: logits = h @ V^T + V_b ----------------
// grid (63, 64): 128x128 output block, K = 128 (padded), 8 waves x 8 N-tiles.
__global__ __launch_bounds__(256) void logits_gemm_kernel(
    const __bf16* __restrict__ Hhi, const __bf16* __restrict__ Hlo,
    const __bf16* __restrict__ Vhi, const __bf16* __restrict__ Vlo,
    const float* __restrict__ Vb, float* __restrict__ out)
{
    const int tid  = threadIdx.x;
    const int wave = tid >> 5, lane = tid & 31;
    const int kh   = lane >> 4, l16 = lane & 15;
    const int row_base = blockIdx.y * 128;
    const int ncb      = blockIdx.x * 128;
    const int mrow     = row_base + wave * 16 + l16;

    v8f acc[8] = {};

#pragma unroll
    for (int kt = 0; kt < 4; ++kt) {
        const int k0 = kt * 32;
        const size_t ao = (size_t)mrow * KPAD_ + k0 + kh * 16;
        ABf ah, al;
        ah.q[0] = *(const u32x4*)(Hhi + ao);
        ah.q[1] = *(const u32x4*)(Hhi + ao + 8);
        al.q[0] = *(const u32x4*)(Hlo + ao);
        al.q[1] = *(const u32x4*)(Hlo + ao + 8);
#pragma unroll
        for (int t = 0; t < 8; ++t) {
            const int n = ncb + t * 16 + l16;
            const size_t bo = (size_t)n * KPAD_ + k0 + kh * 16;
            ABf bh, bl;
            bh.v = *(const v16bf*)(Vhi + bo);
            bl.v = *(const v16bf*)(Vlo + bo);
            acc[t] = wmma3(ah, al, bh, bl, acc[t]);
        }
    }

#pragma unroll
    for (int t = 0; t < 8; ++t) {
        const int col = ncb + t * 16 + l16;
        if (col < V_) {
            const float bias = Vb[col];
#pragma unroll
            for (int r = 0; r < 8; ++r) {
                const int row = row_base + wave * 16 + kh * 8 + r;
                out[(size_t)row * V_ + col] = acc[t][r] + bias;
            }
        }
    }
}

// ---------------- log-softmax: online (max, sumexp) per row, then subtract ----------------
__global__ __launch_bounds__(256) void lse_kernel(
    const float* __restrict__ logits, float* __restrict__ lse)
{
    __shared__ float sm[256], ss[256];
    const int b = blockIdx.x, tid = threadIdx.x;
    const float* row = logits + (size_t)b * V_;
    float m = -3.402823e38f, s = 0.f;
    for (int i = tid; i < V_; i += 256) {
        float v = row[i];
        if (v > m) { s = s * __expf(m - v) + 1.f; m = v; }
        else       { s += __expf(v - m); }
    }
    sm[tid] = m; ss[tid] = s;
    __syncthreads();
    for (int off = 128; off > 0; off >>= 1) {
        if (tid < off) {
            float m1 = sm[tid], m2 = sm[tid + off];
            float mm = fmaxf(m1, m2);
            ss[tid] = ss[tid] * __expf(m1 - mm) + ss[tid + off] * __expf(m2 - mm);
            sm[tid] = mm;
        }
        __syncthreads();
    }
    if (tid == 0) lse[b] = sm[0] + __logf(ss[0]);
}

__global__ __launch_bounds__(256) void lsub_kernel(
    float* __restrict__ out, const float* __restrict__ lse)
{
    int col = blockIdx.x * 256 + threadIdx.x;
    int row = blockIdx.y;
    if (col < V_) out[(size_t)row * V_ + col] -= lse[row];
}

// ---------------- launch ----------------
extern "C" void kernel_launch(void* const* d_in, const int* in_sizes, int n_in,
                              void* d_out, int out_size, void* d_ws, size_t ws_size,
                              hipStream_t stream)
{
    (void)in_sizes; (void)n_in; (void)out_size; (void)ws_size;
    const float* x      = (const float*)d_in[0];
    const float* hidden = (const float*)d_in[1];
    const float* c_in   = (const float*)d_in[2];
    const float* Ui_w = (const float*)d_in[3];
    const float* Uf_w = (const float*)d_in[4];
    const float* Uo_w = (const float*)d_in[5];
    const float* Ug_w = (const float*)d_in[6];
    const float* Wi_w = (const float*)d_in[7];
    const float* Wf_w = (const float*)d_in[8];
    const float* Wo_w = (const float*)d_in[9];
    const float* Wg_w = (const float*)d_in[10];
    const float* V_w  = (const float*)d_in[11];
    const float* Ui_b = (const float*)d_in[12];
    const float* Uf_b = (const float*)d_in[13];
    const float* Uo_b = (const float*)d_in[14];
    const float* Ug_b = (const float*)d_in[15];
    const float* Wi_b = (const float*)d_in[16];
    const float* Wf_b = (const float*)d_in[17];
    const float* Wo_b = (const float*)d_in[18];
    const float* Wg_b = (const float*)d_in[19];
    const float* V_b  = (const float*)d_in[20];

    char* ws = (char*)d_ws;
    __bf16* Uhi = (__bf16*)(ws + OFF_UHI);
    __bf16* Ulo = (__bf16*)(ws + OFF_ULO);
    __bf16* Whi = (__bf16*)(ws + OFF_WHI);
    __bf16* Wlo = (__bf16*)(ws + OFF_WLO);
    __bf16* Vhi = (__bf16*)(ws + OFF_VHI);
    __bf16* Vlo = (__bf16*)(ws + OFF_VLO);
    __bf16* Hhi = (__bf16*)(ws + OFF_HHI);
    __bf16* Hlo = (__bf16*)(ws + OFF_HLO);
    float* gates = (float*)(ws + OFF_GATES);
    float* lse   = (float*)(ws + OFF_LSE);

    float* out   = (float*)d_out;
    float* h_out = out + (size_t)B_ * V_;
    float* c_out = h_out + (size_t)B_ * H_;

    pack_U_kernel<<<dim3(12500), dim3(256), 0, stream>>>(Ui_w, Uf_w, Uo_w, Ug_w, Uhi, Ulo);
    pack_W_kernel<<<dim3(200),   dim3(256), 0, stream>>>(Wi_w, Wf_w, Wo_w, Wg_w, Whi, Wlo);
    pack_V_kernel<<<dim3(4032),  dim3(256), 0, stream>>>(V_w, Vhi, Vlo);

    gates_gemm_kernel<<<dim3(5, 64), dim3(256), 0, stream>>>(
        x, hidden, Uhi, Ulo, Whi, Wlo, gates);

    lstm_kernel<<<dim3(4096), dim3(256), 0, stream>>>(
        gates, c_in, Ui_b, Uf_b, Uo_b, Ug_b, Wi_b, Wf_b, Wo_b, Wg_b,
        h_out, c_out, Hhi, Hlo);

    logits_gemm_kernel<<<dim3(63, 64), dim3(256), 0, stream>>>(
        Hhi, Hlo, Vhi, Vlo, V_b, out);

    lse_kernel<<<dim3(8192), dim3(256), 0, stream>>>(out, lse);
    lsub_kernel<<<dim3(32, 8192), dim3(256), 0, stream>>>(out, lse);
}
